// LightGCN_63264868270398
// MI455X (gfx1250) — matrix operand order repaired and checked by reading, not compiled, not run on hardware
//
#include <hip/hip_runtime.h>
#include <stdint.h>

#define N_USERS 12288
#define N_ITEMS 4096
#define NN      16384            // total nodes
#define DIM     64               // embedding dim
#define WPR     (NN / 32)        // 512 bit-words per adjacency row

typedef __attribute__((ext_vector_type(16))) __bf16 v16bf;
typedef __attribute__((ext_vector_type(8)))  float  v8f;

// ---- helpers -------------------------------------------------------------

// two adjacency bits -> packed {bf16,bf16} of 0.0 / 1.0   (bf16 1.0 = 0x3F80)
__device__ __forceinline__ uint32_t pk2(uint32_t b) {
  return ((b & 1u) ? 0x3F80u : 0u) | ((b & 2u) ? 0x3F800000u : 0u);
}

__device__ __forceinline__ uint16_t f32_to_bf16_rne(float f) {
  uint32_t u = __float_as_uint(f);
  return (uint16_t)((u + 0x7FFFu + ((u >> 16) & 1u)) >> 16);
}

// ---- setup kernels -------------------------------------------------------

__global__ void zero_bits_kernel(uint4* __restrict__ p, int n4) {
  int i = blockIdx.x * blockDim.x + threadIdx.x;
  int stride = gridDim.x * blockDim.x;
  uint4 z = {0u, 0u, 0u, 0u};
  for (; i < n4; i += stride) p[i] = z;
}

__global__ void set_bits_kernel(const int* __restrict__ ui, const int* __restrict__ ii,
                                uint32_t* __restrict__ bits, int ne) {
  int i = blockIdx.x * blockDim.x + threadIdx.x;
  int stride = gridDim.x * blockDim.x;
  for (; i < ne; i += stride) {
    int u = ui[i];
    int v = ii[i] + N_USERS;
    atomicOr(&bits[(size_t)u * WPR + (v >> 5)], 1u << (v & 31));
    atomicOr(&bits[(size_t)v * WPR + (u >> 5)], 1u << (u & 31));
  }
}

// one wave32 per node row: popcount 512 words -> d^{-1/2}
__global__ __launch_bounds__(256) void degree_kernel(const uint32_t* __restrict__ bits,
                                                     float* __restrict__ dinv) {
  int row  = (blockIdx.x * 256 + threadIdx.x) >> 5;
  int lane = threadIdx.x & 31;
  const uint32_t* rp = bits + (size_t)row * WPR;
  int s = 0;
  #pragma unroll
  for (int i = 0; i < WPR / 32; ++i) s += __popc(rp[lane + i * 32]);
  for (int off = 16; off; off >>= 1) s += __shfl_xor(s, off, 32);
  if (lane == 0) dinv[row] = rsqrtf((float)s + 1e-7f);
}

// out = e (f32 accumulator), yt0[d][node] = bf16(dinv[node] * e[node][d])
__global__ void init_kernel(const float* __restrict__ ue, const float* __restrict__ ie,
                            const float* __restrict__ dinv,
                            uint16_t* __restrict__ yt0, float* __restrict__ out) {
  int idx = blockIdx.x * blockDim.x + threadIdx.x;
  if (idx >= NN * DIM) return;
  int node = idx / DIM, d = idx % DIM;
  float e = (node < N_USERS) ? ue[idx] : ie[idx - N_USERS * DIM];
  out[idx] = e;
  yt0[d * NN + node] = f32_to_bf16_rne(dinv[node] * e);
}

__global__ void finalize_kernel(float* __restrict__ out, int n) {
  int idx = blockIdx.x * blockDim.x + threadIdx.x;
  if (idx < n) out[idx] *= 0.25f;
}

// ---- WMMA SpMM: Z = B_bits @ Yt^T ; out += Dinv*Z ; ytout = Dinv^2*Z -----
// 256 threads = 8 wave32, block covers 64 rows.
// Wave w: row-tile (w&3)*16, N-half (w>>2) -> 2 accumulator tiles per wave.
// Register double-buffering: next K-chunk (Yt slice + adjacency bits) is
// loaded into VGPRs while the current chunk's 8 WMMAs execute (L2-resident).
__global__ __launch_bounds__(256) void spmm_kernel(
    const uint32_t* __restrict__ bits, const float* __restrict__ dinv,
    const uint16_t* __restrict__ ytin, uint16_t* __restrict__ ytout,
    float* __restrict__ out) {
  // K-slice of Yt, [n=0..63][kk=0..127], row stride padded to 136 (272B) for banks
  __shared__ __align__(16) uint16_t Ys[64][136];

  const int tid   = threadIdx.x;
  const int lane  = tid & 31;
  const int w     = tid >> 5;          // wave id 0..7
  const int mtile = w & 3;             // which 16-row tile
  const int nhalf = w >> 2;            // which pair of 16-col tiles
  const int m     = lane & 15;
  const int khalf = lane >> 4;         // 0: lanes 0-15, 1: lanes 16-31
  const int rowbase = blockIdx.x * 64;
  const int wrow    = rowbase + mtile * 16;
  const int myrow   = wrow + m;        // adjacency row this lane fetches
  const int sn = tid >> 2;             // LDS staging: Yt row 0..63
  const int sq = tid & 3;              //              quarter of 256B row

  v8f acc[2] = {v8f{}, v8f{}};

  // ---- prologue: prefetch chunk 0 into registers
  uint4 stage[4];
  #pragma unroll
  for (int c = 0; c < 4; ++c)
    stage[c] = *(const uint4*)(ytin + (size_t)sn * NN + sq * 32 + c * 8);
  uint4 bw = *(const uint4*)(bits + (size_t)myrow * WPR);

  for (int k0 = 0; k0 < NN; k0 += 128) {
    __syncthreads();                       // previous chunk's compute done
    #pragma unroll
    for (int c = 0; c < 4; ++c)
      *(uint4*)(&Ys[sn][sq * 32 + c * 8]) = stage[c];
    const uint4 cur = bw;
    __syncthreads();                       // slice visible to all waves

    if (k0 + 128 < NN) {                   // uniform; prefetch next chunk
      #pragma unroll
      for (int c = 0; c < 4; ++c)
        stage[c] = *(const uint4*)(ytin + (size_t)sn * NN + (k0 + 128) + sq * 32 + c * 8);
      bw = *(const uint4*)(bits + (size_t)myrow * WPR + ((k0 + 128) >> 5));
    }

    const uint32_t wds[4] = {cur.x, cur.y, cur.z, cur.w};
    #pragma unroll
    for (int kc = 0; kc < 4; ++kc) {
      uint32_t word = wds[kc];
      // ISA 16-bit A 16x32 layout: lanes 0-15 hold K 0-7 (v0-3) & 16-23 (v4-7),
      // lanes 16-31 hold K 8-15 & 24-31
      uint32_t blo = khalf ? ((word >> 8)  & 0xFFu) : (word & 0xFFu);
      uint32_t bhi = khalf ? ((word >> 24) & 0xFFu) : ((word >> 16) & 0xFFu);
      union { uint32_t u[8]; v16bf v; } A;
      #pragma unroll
      for (int j = 0; j < 4; ++j) {
        A.u[j]     = pk2(blo >> (2 * j));
        A.u[4 + j] = pk2(bhi >> (2 * j));
      }
      #pragma unroll
      for (int t = 0; t < 2; ++t) {
        // B 32x16 layout: lane = N col, same K order as A (lanes 0-15 -> K 0-7 & 16-23,
        // 16-31 -> K 8-15 & 24-31): 8 bf16 from Ys[n][kc*32 + khalf*8 ..] and 8 from +16
        const uint16_t* bp = &Ys[(nhalf * 2 + t) * 16 + m][kc * 32 + khalf * 8];
        union { uint4 q[2]; v16bf v; } B;
        B.q[0] = *(const uint4*)bp;
        B.q[1] = *(const uint4*)(bp + 16);
        acc[t] = __builtin_amdgcn_wmma_f32_16x16x32_bf16(
            false, A.v, false, B.v, (short)0, acc[t], false, false);
      }
    }
  }

  // Epilogue: C/D layout -> element r is (M = r + 8*khalf, N = lane&15)
  float dv[8];
  #pragma unroll
  for (int r = 0; r < 8; ++r) dv[r] = dinv[wrow + khalf * 8 + r];

  #pragma unroll
  for (int t = 0; t < 2; ++t) {
    int col = (nhalf * 2 + t) * 16 + m;
    #pragma unroll
    for (int r = 0; r < 8; ++r) {
      int row  = wrow + khalf * 8 + r;
      float z  = acc[t][r];
      float x  = dv[r] * z;                       // this layer's embedding
      out[row * DIM + col] += x;                  // LightGCN running sum
      ytout[col * NN + row] = f32_to_bf16_rne(dv[r] * x);  // Dinv^2*Z next layer
    }
  }
}

// ---- launch --------------------------------------------------------------

extern "C" void kernel_launch(void* const* d_in, const int* in_sizes, int n_in,
                              void* d_out, int out_size, void* d_ws, size_t ws_size,
                              hipStream_t stream) {
  const int*   ui = (const int*)d_in[0];
  const int*   ii = (const int*)d_in[1];
  const float* ue = (const float*)d_in[2];
  const float* ie = (const float*)d_in[3];
  float* out = (float*)d_out;
  const int ne = in_sizes[0];

  // workspace: 32MB bit adjacency | 64KB dinv | 2MB yt0 | 2MB yt1
  char* ws = (char*)d_ws;
  uint32_t* bits = (uint32_t*)ws;
  size_t off = (size_t)NN * WPR * sizeof(uint32_t);
  float* dinv = (float*)(ws + off);
  off += (size_t)NN * sizeof(float);
  off = (off + 255) & ~(size_t)255;
  uint16_t* yt0 = (uint16_t*)(ws + off);
  off += (size_t)NN * DIM * sizeof(uint16_t);
  off = (off + 255) & ~(size_t)255;
  uint16_t* yt1 = (uint16_t*)(ws + off);

  zero_bits_kernel<<<2048, 256, 0, stream>>>((uint4*)bits, NN * WPR / 4);
  set_bits_kernel<<<1024, 256, 0, stream>>>(ui, ii, bits, ne);
  degree_kernel<<<NN / 8, 256, 0, stream>>>(bits, dinv);
  init_kernel<<<(NN * DIM + 255) / 256, 256, 0, stream>>>(ue, ie, dinv, yt0, out);

  spmm_kernel<<<NN / 64, 256, 0, stream>>>(bits, dinv, yt0, yt1, out);
  spmm_kernel<<<NN / 64, 256, 0, stream>>>(bits, dinv, yt1, yt0, out);
  spmm_kernel<<<NN / 64, 256, 0, stream>>>(bits, dinv, yt0, yt1, out);

  finalize_kernel<<<(NN * DIM + 255) / 256, 256, 0, stream>>>(out, NN * DIM);
}
